// ModelLstm_25245817766462
// MI455X (gfx1250) — matrix operand order repaired and checked by reading, not compiled, not run on hardware
//
#include <hip/hip_runtime.h>
#include <hip/hip_bf16.h>

#define NODES   1024
#define GATES   4096
#define SEQT    2048
#define NLAYERS 5
#define INDIM   30
#define OUTDIM  12
#define SLOPE   0.01f
#define RECBLOCKS 32u

typedef __attribute__((ext_vector_type(16))) __bf16         v16bf;
typedef __attribute__((ext_vector_type(8)))  float          v8f;
typedef __attribute__((ext_vector_type(16))) unsigned short v16u;
typedef __attribute__((ext_vector_type(8)))  unsigned short v8u;

__device__ __forceinline__ unsigned short f32_to_bf16_bits(float f) {
    unsigned u = __float_as_uint(f);
    u += 0x7FFFu + ((u >> 16) & 1u);       // round-to-nearest-even
    return (unsigned short)(u >> 16);
}
__device__ __forceinline__ float lrelu(float x)    { return x > 0.f ? x : SLOPE * x; }
__device__ __forceinline__ float sigmoidf(float x) { return 1.f / (1.f + expf(-x)); }

// ---------------- device-wide split barrier (32 co-resident blocks) --------
__device__ __forceinline__ void grid_sync(unsigned* bar, unsigned nblocks) {
    __threadfence();
    __syncthreads();
    if (threadIdx.x == 0) {
        unsigned g = atomicAdd(&bar[1], 0u);
        if (atomicAdd(&bar[0], 1u) + 1u == nblocks) {
            atomicExch(&bar[0], 0u);
            __threadfence();
            atomicAdd(&bar[1], 1u);
        } else {
            while (atomicAdd(&bar[1], 0u) == g) { __builtin_amdgcn_s_sleep(2); }
        }
    }
    __syncthreads();
    __threadfence();
}

// ---------------- kernel 0: zero barrier state -----------------------------
__global__ void init_bar_kernel(unsigned* bar) {
    if (threadIdx.x < 2) bar[threadIdx.x] = 0u;
}

// ---------------- kernel 1: concat + W1 + leaky_relu -----------------------
__global__ __launch_bounds__(256) void fuse_input_kernel(
    const float* __restrict__ sim, const float* __restrict__ realv,
    const float* __restrict__ act, const float* __restrict__ W1,
    const float* __restrict__ b1, float* __restrict__ x) {
    int gid = blockIdx.x * 256 + threadIdx.x;
    if (gid >= SEQT * NODES) return;
    int t = gid >> 10, n = gid & (NODES - 1);
    const float* w  = W1 + n * INDIM;
    const float* ps = sim   + t * 12;
    const float* pr = realv + t * 12;
    const float* pa = act   + t * 6;
    float s = b1[n];
    #pragma unroll
    for (int i = 0; i < 12; ++i) s += ps[i] * w[i];
    #pragma unroll
    for (int i = 0; i < 12; ++i) s += pr[i] * w[12 + i];
    #pragma unroll
    for (int i = 0; i < 6;  ++i) s += pa[i] * w[24 + i];
    x[gid] = lrelu(s);
}

// ---------------- kernel 2: fp32 -> bf16 -----------------------------------
__global__ __launch_bounds__(256) void cvt_bf16_kernel(
    const float* __restrict__ src, unsigned short* __restrict__ dst, int n) {
    int gid = blockIdx.x * 256 + threadIdx.x;
    if (gid < n) dst[gid] = f32_to_bf16_bits(src[gid]);
}

// ---------------- kernel 3: WMMA GEMM  xproj = X * W_ih^T + (b_ih+b_hh) ----
// X: [2048,1024] bf16 row-major, W: [4096,1024] bf16 row-major, D: [2048,4096] f32
// One wave -> 32(M) x 64(N) tile: 8 WMMAs per k-step vs 6 fragment loads
// (2 A + 4 B), so B fragments amortize over two M-tiles. The 8 waves of a
// block share nt (same B lines -> WGP$ hits) with consecutive mt.
__global__ __launch_bounds__(256) void gemm_xproj_kernel(
    const unsigned short* __restrict__ A16, const unsigned short* __restrict__ W16,
    const float* __restrict__ bih, const float* __restrict__ bhh,
    float* __restrict__ D) {
    const int lane  = threadIdx.x & 31;
    const int widx  = threadIdx.x >> 5;
    const int id    = blockIdx.x * 8 + widx;        // 0..4095
    const int mt    = id & 63;                      // 64 M-supertiles (32 rows)
    const int nt    = id >> 6;                      // 64 N-supertiles (64 cols)
    const int m0    = mt * 32;
    const int n0    = nt * 64;
    const int row   = lane & 15;
    const int khalf = lane >> 4;                    // 0 or 1

    const unsigned short* arow0 = A16 + (size_t)(m0 + row) * NODES + khalf * 16;
    const unsigned short* arow1 = arow0 + (size_t)16 * NODES;
    const unsigned short* w0 = W16 + (size_t)(n0 +  0 + row) * NODES + khalf * 16;
    const unsigned short* w1 = W16 + (size_t)(n0 + 16 + row) * NODES + khalf * 16;
    const unsigned short* w2 = W16 + (size_t)(n0 + 32 + row) * NODES + khalf * 16;
    const unsigned short* w3 = W16 + (size_t)(n0 + 48 + row) * NODES + khalf * 16;

    v8f acc[2][4];
    #pragma unroll
    for (int mi = 0; mi < 2; ++mi)
        #pragma unroll
        for (int nj = 0; nj < 4; ++nj) acc[mi][nj] = {};

    for (int kk = 0; kk < NODES; kk += 32) {
        union { v16u u; v16bf b; } a0, a1, b0, b1v, b2v, b3v;
        {
            v8u lo = *(const v8u*)(arow0 + kk);
            v8u hi = *(const v8u*)(arow0 + kk + 8);
            #pragma unroll
            for (int i = 0; i < 8; ++i) { a0.u[i] = lo[i]; a0.u[8 + i] = hi[i]; }
        }
        {
            v8u lo = *(const v8u*)(arow1 + kk);
            v8u hi = *(const v8u*)(arow1 + kk + 8);
            #pragma unroll
            for (int i = 0; i < 8; ++i) { a1.u[i] = lo[i]; a1.u[8 + i] = hi[i]; }
        }
        b0.u  = *(const v16u*)(w0 + kk);
        b1v.u = *(const v16u*)(w1 + kk);
        b2v.u = *(const v16u*)(w2 + kk);
        b3v.u = *(const v16u*)(w3 + kk);
        __builtin_prefetch(arow0 + kk + 64, 0, 1);
        __builtin_prefetch(arow1 + kk + 64, 0, 1);
        __builtin_prefetch(w0 + kk + 64, 0, 1);
        acc[0][0] = __builtin_amdgcn_wmma_f32_16x16x32_bf16(false, a0.b, false, b0.b,  (short)0, acc[0][0], false, false);
        acc[0][1] = __builtin_amdgcn_wmma_f32_16x16x32_bf16(false, a0.b, false, b1v.b, (short)0, acc[0][1], false, false);
        acc[0][2] = __builtin_amdgcn_wmma_f32_16x16x32_bf16(false, a0.b, false, b2v.b, (short)0, acc[0][2], false, false);
        acc[0][3] = __builtin_amdgcn_wmma_f32_16x16x32_bf16(false, a0.b, false, b3v.b, (short)0, acc[0][3], false, false);
        acc[1][0] = __builtin_amdgcn_wmma_f32_16x16x32_bf16(false, a1.b, false, b0.b,  (short)0, acc[1][0], false, false);
        acc[1][1] = __builtin_amdgcn_wmma_f32_16x16x32_bf16(false, a1.b, false, b1v.b, (short)0, acc[1][1], false, false);
        acc[1][2] = __builtin_amdgcn_wmma_f32_16x16x32_bf16(false, a1.b, false, b2v.b, (short)0, acc[1][2], false, false);
        acc[1][3] = __builtin_amdgcn_wmma_f32_16x16x32_bf16(false, a1.b, false, b3v.b, (short)0, acc[1][3], false, false);
    }

    const int ncol  = lane & 15;
    const int rbase = (lane >> 4) * 8;   // C layout: lanes 16-31 hold rows 8-15
    #pragma unroll
    for (int nj = 0; nj < 4; ++nj) {
        int col = n0 + nj * 16 + ncol;
        float bias = bih[col] + bhh[col];
        #pragma unroll
        for (int mi = 0; mi < 2; ++mi) {
            #pragma unroll
            for (int vg = 0; vg < 8; ++vg) {
                D[(size_t)(m0 + mi * 16 + rbase + vg) * GATES + col] = acc[mi][nj][vg] + bias;
            }
        }
    }
}

// ---------------- kernel 4: per-layer state init ---------------------------
__global__ __launch_bounds__(256) void init_state_kernel(
    const float* __restrict__ h0, const float* __restrict__ c0, int layer,
    unsigned short* __restrict__ h16, float* __restrict__ cbuf) {
    int j = blockIdx.x * 256 + threadIdx.x;
    if (j < NODES) {
        h16[j]  = f32_to_bf16_bits(h0[layer * NODES + j]);
        cbuf[j] = c0[layer * NODES + j];
    }
}

// ---------------- kernel 5: cooperative LSTM recurrence --------------------
// 32 blocks x 256 threads = 256 waves; wave w owns gate columns [16w,16w+16).
// Matvec h(1x1024) * Whh^T via WMMA with h in A-fragment row 0 (rows 1-15 = 0):
// bandwidth-bound on B (L2-resident 8MB Whh), WMMA streams B at full rate.
__global__ __launch_bounds__(256) void lstm_rec_kernel(
    const unsigned short* __restrict__ Whh16, const float* __restrict__ xproj,
    unsigned short* h16, float* cbuf, float* gates,
    float* __restrict__ y, unsigned* bar) {
    const int lane  = threadIdx.x & 31;
    const int widx  = threadIdx.x >> 5;
    const int n0    = (blockIdx.x * 8 + widx) * 16;
    const int col   = n0 + (lane & 15);
    const int khalf = lane >> 4;
    const unsigned short* wrow = Whh16 + (size_t)col * NODES + khalf * 16;
    const int gtid  = blockIdx.x * 256 + threadIdx.x;
    const bool aldr = (lane == 0) | (lane == 16);
    const int  koff = khalf * 16;

    for (int t = 0; t < SEQT; ++t) {
        // ---- phase 1: gates = h @ Whh^T (+ precomputed xproj incl. biases)
        v8f acc = {};
        #pragma unroll 4
        for (int kk = 0; kk < NODES; kk += 32) {
            union { v16u u; v16bf b; } a, bb;
            #pragma unroll
            for (int i = 0; i < 16; ++i) a.u[i] = 0;
            if (aldr) {
                v8u lo = *(const v8u*)(h16 + kk + koff);
                v8u hi = *(const v8u*)(h16 + kk + 8 + koff);
                #pragma unroll
                for (int i = 0; i < 8; ++i) { a.u[i] = lo[i]; a.u[8 + i] = hi[i]; }
            }
            bb.u = *(const v16u*)(wrow + kk);
            acc = __builtin_amdgcn_wmma_f32_16x16x32_bf16(false, a.b, false, bb.b, (short)0, acc, false, false);
        }
        if (lane < 16) gates[col] = acc[0] + xproj[(size_t)t * GATES + col];
        grid_sync(bar, RECBLOCKS);

        // ---- phase 2: elementwise cell update (first 1024 threads)
        if (gtid < NODES) {
            int j = gtid;
            float gi = sigmoidf(gates[j]);
            float gf = sigmoidf(gates[j + NODES]);
            float gg = tanhf(gates[j + 2 * NODES]);
            float go = sigmoidf(gates[j + 3 * NODES]);
            float cv = gf * cbuf[j] + gi * gg;
            cbuf[j] = cv;
            float hv = go * tanhf(cv);
            h16[j] = f32_to_bf16_bits(hv);
            y[(size_t)t * NODES + j] = hv;
        }
        grid_sync(bar, RECBLOCKS);
    }
}

// ---------------- kernel 6: out = leaky_relu(y) @ W2^T + b2 ----------------
__global__ __launch_bounds__(256) void out_proj_kernel(
    const float* __restrict__ y, const float* __restrict__ W2,
    const float* __restrict__ b2, float* __restrict__ out) {
    int gid  = blockIdx.x * 256 + threadIdx.x;
    int wave = gid >> 5, lane = gid & 31;
    if (wave >= SEQT * OUTDIM) return;
    int t = wave / OUTDIM, o = wave % OUTDIM;
    const float* yr = y  + (size_t)t * NODES;
    const float* wr = W2 + (size_t)o * NODES;
    float s = 0.f;
    for (int k = lane; k < NODES; k += 32) s += lrelu(yr[k]) * wr[k];
    #pragma unroll
    for (int off = 16; off; off >>= 1) s += __shfl_down(s, off, 32);
    if (lane == 0) out[wave] = s + b2[o];
}

// ---------------- host orchestration ---------------------------------------
extern "C" void kernel_launch(void* const* d_in, const int* in_sizes, int n_in,
                              void* d_out, int out_size, void* d_ws, size_t ws_size,
                              hipStream_t stream) {
    const float* sim   = (const float*)d_in[0];
    const float* realv = (const float*)d_in[1];
    const float* act   = (const float*)d_in[2];
    const float* W1    = (const float*)d_in[3];
    const float* b1    = (const float*)d_in[4];
    const float* W_ih  = (const float*)d_in[5];
    const float* W_hh  = (const float*)d_in[6];
    const float* b_ih  = (const float*)d_in[7];
    const float* b_hh  = (const float*)d_in[8];
    const float* W2    = (const float*)d_in[9];
    const float* b2    = (const float*)d_in[10];
    const float* h0    = (const float*)d_in[11];
    const float* c0    = (const float*)d_in[12];
    float* out = (float*)d_out;

    char* ws = (char*)d_ws;
    size_t off = 0;
    auto alloc = [&](size_t bytes) { size_t o = off; off = (off + bytes + 255) & ~(size_t)255; return o; };
    float*          bufA   = (float*)         (ws + alloc((size_t)SEQT * NODES * 4));
    float*          bufB   = (float*)         (ws + alloc((size_t)SEQT * NODES * 4));
    unsigned short* a16    = (unsigned short*)(ws + alloc((size_t)SEQT * NODES * 2));
    unsigned short* wih16  = (unsigned short*)(ws + alloc((size_t)GATES * NODES * 2));
    unsigned short* whh16  = (unsigned short*)(ws + alloc((size_t)GATES * NODES * 2));
    float*          xproj  = (float*)         (ws + alloc((size_t)SEQT * GATES * 4));
    unsigned short* h16    = (unsigned short*)(ws + alloc((size_t)NODES * 2));
    float*          cbuf   = (float*)         (ws + alloc((size_t)NODES * 4));
    float*          gates  = (float*)         (ws + alloc((size_t)GATES * 4));
    unsigned*       bar    = (unsigned*)      (ws + alloc(256));

    init_bar_kernel<<<1, 64, 0, stream>>>(bar);
    fuse_input_kernel<<<(SEQT * NODES) / 256, 256, 0, stream>>>(sim, realv, act, W1, b1, bufA);

    float* cur = bufA;
    float* nxt = bufB;
    for (int l = 0; l < NLAYERS; ++l) {
        const int NW = GATES * NODES;
        cvt_bf16_kernel<<<NW / 256, 256, 0, stream>>>(W_ih + (size_t)l * NW, wih16, NW);
        cvt_bf16_kernel<<<NW / 256, 256, 0, stream>>>(W_hh + (size_t)l * NW, whh16, NW);
        cvt_bf16_kernel<<<(SEQT * NODES) / 256, 256, 0, stream>>>(cur, a16, SEQT * NODES);
        gemm_xproj_kernel<<<512, 256, 0, stream>>>(a16, wih16,
                                                   b_ih + (size_t)l * GATES,
                                                   b_hh + (size_t)l * GATES, xproj);
        init_state_kernel<<<4, 256, 0, stream>>>(h0, c0, l, h16, cbuf);
        lstm_rec_kernel<<<RECBLOCKS, 256, 0, stream>>>(whh16, xproj, h16, cbuf, gates, nxt, bar);
        float* tmp = cur; cur = nxt; nxt = tmp;
    }

    out_proj_kernel<<<(SEQT * OUTDIM * 32) / 256, 256, 0, stream>>>(cur, W2, b2, out);
}
